// TransformAggregator_20444044329183
// MI455X (gfx1250) — matrix operand order repaired and checked by reading, not compiled, not run on hardware
//
#include <hip/hip_runtime.h>

// Problem constants (match reference)
#define B_    32
#define N_    512
#define D_    512
#define H_    4
#define DH_   128
#define FF_   2048
#define NPAD  544          // 513 padded to multiple of 32 (17 k-chunks / 34 m-tiles)
#define LN_EPS 1e-5f

typedef __attribute__((ext_vector_type(16))) __bf16 v16bf;
typedef __attribute__((ext_vector_type(8)))  __bf16 v8bf;
typedef __attribute__((ext_vector_type(8)))  float  v8f;

// ---------- small helpers ----------
static __device__ inline unsigned short f2bf(float f) {
  unsigned int u = __float_as_uint(f);
  u += 0x7FFFu + ((u >> 16) & 1u);          // round-to-nearest-even
  return (unsigned short)(u >> 16);
}
static __device__ inline float bf2f(unsigned short h) {
  return __uint_as_float(((unsigned int)h) << 16);
}

static __device__ inline float block_reduce_sum(float v, float* sm) {
  int t = threadIdx.x;
  sm[t] = v; __syncthreads();
  for (int s = 128; s > 0; s >>= 1) {
    if (t < s) sm[t] += sm[t + s];
    __syncthreads();
  }
  float r = sm[0]; __syncthreads();
  return r;
}
static __device__ inline float block_reduce_max(float v, float* sm) {
  int t = threadIdx.x;
  sm[t] = v; __syncthreads();
  for (int s = 128; s > 0; s >>= 1) {
    if (t < s) sm[t] = fmaxf(sm[t], sm[t + s]);
    __syncthreads();
  }
  float r = sm[0]; __syncthreads();
  return r;
}

// ---------- WMMA fragment loaders (bf16, 16x16x32) ----------
// A matrix 16x32 (M x K), 16-bit layout:
//  lanes 0-15 : row M=lane, K = {0..7} in VGPR0-3, K = {16..23} in VGPR4-7
//  lanes 16-31: row M=lane-16, K = {8..15}, {24..31}
static __device__ inline v16bf load_a_frag(const unsigned short* Abase, int lda,
                                           int lane, int kk) {
  int row = lane & 15, half = lane >> 4;
  const unsigned short* p = Abase + (size_t)row * lda + kk * 32 + half * 8;
  v8bf lo = *(const v8bf*)p;
  v8bf hi = *(const v8bf*)(p + 16);
  return __builtin_shufflevector(lo, hi, 0,1,2,3,4,5,6,7,8,9,10,11,12,13,14,15);
}

// B matrix 32x16 (K x N), stored COLUMN-major (Bcm[n*ldcol + k]):
//  lanes 0-15 : col N=lane, K = kk*32 + {0..7, 16..23} ; lanes 16-31: col N=lane-16, K = +{8..15, 24..31}
// Unconditional (all 16 columns must be addressable; small B operands are padded).
static __device__ inline v16bf load_b_frag_cm(const unsigned short* Bcm, int ldcol,
                                              int lane, int kk, int ncolbase) {
  int n = lane & 15, half = lane >> 4;
  const unsigned short* p = Bcm + (size_t)(ncolbase + n) * ldcol + kk * 32 + half * 8;
  v8bf lo = *(const v8bf*)p;
  v8bf hi = *(const v8bf*)(p + 16);
  return __builtin_shufflevector(lo, hi, 0,1,2,3,4,5,6,7,8,9,10,11,12,13,14,15);
}

static __device__ inline v8f wmma_loop(const unsigned short* A, int lda,
                                       const unsigned short* Bcm, int ldcol,
                                       int ncolbase, int ksteps, int lane) {
  v8f c = {};
  for (int kk = 0; kk < ksteps; ++kk) {
    v16bf a = load_a_frag(A, lda, lane, kk);
    v16bf b = load_b_frag_cm(Bcm, ldcol, lane, kk, ncolbase);
    c = __builtin_amdgcn_wmma_f32_16x16x32_bf16(false, a, false, b,
                                                (short)0, c, false, false);
  }
  return c;
}

// ---------- kernels ----------

// L2-normalize each token row, then LN1, write bf16 h[b, j, :]
__global__ void k_norm_ln1(const float* __restrict__ x,
                           const float* __restrict__ ln1s,
                           const float* __restrict__ ln1b,
                           unsigned short* __restrict__ h) {
  __shared__ float red[256];
  int r = blockIdx.x;                 // b*N + j
  int b = r >> 9, j = r & 511;
  int t = threadIdx.x;
  const float* xr = x + (size_t)r * D_;
  float a0 = xr[t], a1 = xr[t + 256];
  float ss = block_reduce_sum(a0 * a0 + a1 * a1, red);
  float sx = block_reduce_sum(a0 + a1, red);
  float inv = 1.0f / fmaxf(sqrtf(ss), 1e-12f);
  float mean = sx * inv * (1.0f / D_);
  float var  = ss * inv * inv * (1.0f / D_) - mean * mean;
  float rstd = rsqrtf(var + LN_EPS);
  size_t ob = ((size_t)b * NPAD + j) * D_;
  h[ob + t]       = f2bf((a0 * inv - mean) * rstd * ln1s[t]       + ln1b[t]);
  h[ob + t + 256] = f2bf((a1 * inv - mean) * rstd * ln1s[t + 256] + ln1b[t + 256]);
}

// one block: h_cls = LN1(token); q = h_cls@Wq+bq; m[h,a] = sum_{c in head} Wk[a,c] q[c];
// qb[h] = q_head . bk_head   (mbf rows 4..15 are pre-zeroed by memset)
__global__ void k_prep_small(const float* __restrict__ token,
                             const float* __restrict__ ln1s,
                             const float* __restrict__ ln1b,
                             const float* __restrict__ Wq, const float* __restrict__ bq,
                             const float* __restrict__ Wk, const float* __restrict__ bk,
                             unsigned short* __restrict__ hclsb,
                             unsigned short* __restrict__ mbf,
                             float* __restrict__ qb) {
  __shared__ float red[256];
  __shared__ float hc[512];
  __shared__ float qs[512];
  int t = threadIdx.x;
  float a0 = token[t], a1 = token[t + 256];
  float ss = block_reduce_sum(a0 * a0 + a1 * a1, red);
  float sx = block_reduce_sum(a0 + a1, red);
  float mean = sx * (1.0f / D_);
  float var  = ss * (1.0f / D_) - mean * mean;
  float rstd = rsqrtf(var + LN_EPS);
  float h0  = (a0 - mean) * rstd * ln1s[t]       + ln1b[t];
  float h1v = (a1 - mean) * rstd * ln1s[t + 256] + ln1b[t + 256];
  hc[t] = h0; hc[t + 256] = h1v;
  hclsb[t] = f2bf(h0); hclsb[t + 256] = f2bf(h1v);
  __syncthreads();
  for (int cc = t; cc < 512; cc += 256) {
    float acc = bq[cc];
    for (int a = 0; a < 512; ++a) acc += hc[a] * Wq[(size_t)a * D_ + cc];
    qs[cc] = acc;
  }
  __syncthreads();
  for (int aa = t; aa < 512; aa += 256) {
    for (int hh = 0; hh < H_; ++hh) {
      float acc = 0.0f;
      const float* wr = Wk + (size_t)aa * D_ + hh * DH_;
      const float* qp = qs + hh * DH_;
      for (int c = 0; c < DH_; ++c) acc += wr[c] * qp[c];
      mbf[hh * D_ + aa] = f2bf(acc);
    }
  }
  if (t < H_) {
    float acc = 0.0f;
    for (int c = 0; c < DH_; ++c) acc += qs[t * DH_ + c] * bk[t * DH_ + c];
    qb[t] = acc;
  }
}

// rows 512..543 of each batch: row 512 = h_cls, rest zero
__global__ void k_fill_pad(unsigned short* __restrict__ h,
                           const unsigned short* __restrict__ hclsb) {
  int bx = blockIdx.x;                // B*32 blocks
  int b = bx >> 5, p = bx & 31;
  int t = threadIdx.x;                // 128
  size_t base = ((size_t)b * NPAD + 512 + p) * D_;
  int c = t * 4;
  if (p == 0) {
    for (int i = 0; i < 4; ++i) h[base + c + i] = hclsb[c + i];
  } else {
    for (int i = 0; i < 4; ++i) h[base + c + i] = 0;
  }
}

// transposed f32->bf16 weight convert: dst[n*K + k] = src[k*N + n]
__global__ void k_convT(const float* __restrict__ src, unsigned short* __restrict__ dst,
                        int K, int N) {
  __shared__ float tile[32][33];
  int n0 = blockIdx.x * 32, k0 = blockIdx.y * 32;
  int tx = threadIdx.x, ty = threadIdx.y;   // (32, 8)
  for (int i = 0; i < 4; ++i)
    tile[ty + i * 8][tx] = src[(size_t)(k0 + ty + i * 8) * N + n0 + tx];
  __syncthreads();
  for (int i = 0; i < 4; ++i)
    dst[(size_t)(n0 + ty + i * 8) * K + k0 + tx] = f2bf(tile[tx][ty + i * 8]);
}

__global__ void k_conv(const float* __restrict__ src, unsigned short* __restrict__ dst, int n) {
  for (int i = blockIdx.x * blockDim.x + threadIdx.x; i < n; i += gridDim.x * blockDim.x)
    dst[i] = f2bf(src[i]);
}

// scores[b,h,j] = (h[b,j,:].mbf[h,:] + qb[h]) / sqrt(DH)   via WMMA, one wave per (b, mtile)
__global__ void k_scores(const unsigned short* __restrict__ h,
                         const unsigned short* __restrict__ mbf,
                         const float* __restrict__ qb,
                         float* __restrict__ sc) {
  int bx = blockIdx.x;
  int b = bx / 34, mt = bx % 34;
  int lane = threadIdx.x;
  const unsigned short* A = h + ((size_t)b * NPAD + mt * 16) * D_;
  v8f c = wmma_loop(A, D_, mbf, D_, 0, 16, lane);
  int col = lane & 15, half = lane >> 4;
  if (col < H_) {
    float qv = qb[col];
    float* o = sc + ((size_t)b * H_ + col) * NPAD + mt * 16 + half * 8;
    for (int i = 0; i < 8; ++i) o[i] = (c[i] + qv) * 0.08838834764831845f;
  }
}

// masked softmax in place over sequence (valid: j < mask[b] or j == 512)
__global__ void k_softmax(float* __restrict__ sc, const int* __restrict__ mask) {
  __shared__ float red[256];
  int bx = blockIdx.x;
  int b = bx >> 2, hh = bx & 3;
  int t = threadIdx.x;
  float* p = sc + ((size_t)b * H_ + hh) * NPAD;
  int mk = mask[b];
  float mx = -3.0e38f;
  for (int j = t; j < NPAD; j += 256) {
    bool valid = (j < mk) || (j == 512);
    if (valid) mx = fmaxf(mx, p[j]);
  }
  mx = block_reduce_max(mx, red);
  float sm = 0.0f;
  for (int j = t; j < NPAD; j += 256) {
    bool valid = (j < mk) || (j == 512);
    float e = valid ? expf(p[j] - mx) : 0.0f;
    p[j] = e; sm += e;
  }
  sm = block_reduce_sum(sm, red);
  float inv = 1.0f / sm;
  for (int j = t; j < NPAD; j += 256) p[j] *= inv;
}

// s[b,hh,a] = sum_j attn[b,hh,j] * h[b,j,a]
// Streaming pass double-buffered through LDS with gfx1250 async global->LDS copies.
// Each block: one batch, one 256-column half, one group of 136 rows (17 tiles of 8 rows).
__global__ void k_wsum(const unsigned short* __restrict__ h,
                       const float* __restrict__ attn,
                       float* __restrict__ s) {
  __shared__ unsigned short lbuf[2][2048];   // 2 x (8 rows x 256 cols bf16) = 2 x 4 KB
  int bx = blockIdx.x;                       // B * 2 * 4 = 256 blocks
  int b  = bx >> 3;
  int rem = bx & 7;
  int ch = rem >> 2;                         // column half (0/1)
  int jg = rem & 3;                          // row group of 136
  int t = threadIdx.x;
  int jbase = jg * 136;
  const int NSUB = 17;

  // async-copy lane mapping: thread -> (row 0..7, 16B segment 0..31)
  int lr = t >> 5;
  int lseg = t & 31;
  unsigned ldsbase = (unsigned)(size_t)(&lbuf[0][0]);
  const char* gsrc = (const char*)(h + (size_t)b * NPAD * D_ + ch * 256);

  // prime buffer 0
  {
    unsigned dst = ldsbase + (unsigned)(lr * 512 + lseg * 16);
    const char* src = gsrc + (size_t)(jbase + lr) * (D_ * 2) + lseg * 16;
    asm volatile("global_load_async_to_lds_b128 %0, %1, off"
                 :: "v"(dst), "v"(src) : "memory");
  }
  asm volatile("s_wait_asynccnt 0x0" ::: "memory");
  __syncthreads();

  const float* ab = attn + (size_t)b * H_ * NPAD;
  float acc0 = 0.f, acc1 = 0.f, acc2 = 0.f, acc3 = 0.f;
  for (int c = 0; c < NSUB; ++c) {
    int cur = c & 1;
    if (c + 1 < NSUB) {   // kick off next tile while computing on current
      unsigned dst = ldsbase + (unsigned)(((c + 1) & 1) * 4096 + lr * 512 + lseg * 16);
      const char* src = gsrc + (size_t)(jbase + (c + 1) * 8 + lr) * (D_ * 2) + lseg * 16;
      asm volatile("global_load_async_to_lds_b128 %0, %1, off"
                   :: "v"(dst), "v"(src) : "memory");
    }
    int j0 = jbase + c * 8;
    for (int r = 0; r < 8; ++r) {
      float hv = bf2f(lbuf[cur][r * 256 + t]);
      int j = j0 + r;
      acc0 += ab[0 * NPAD + j] * hv;
      acc1 += ab[1 * NPAD + j] * hv;
      acc2 += ab[2 * NPAD + j] * hv;
      acc3 += ab[3 * NPAD + j] * hv;
    }
    asm volatile("s_wait_asynccnt 0x0" ::: "memory");
    __syncthreads();
  }

  int a = ch * 256 + t;
  float* sp = s + (size_t)b * H_ * D_ + a;
  atomicAdd(sp + 0 * D_, acc0);
  atomicAdd(sp + 1 * D_, acc1);
  atomicAdd(sp + 2 * D_, acc2);
  atomicAdd(sp + 3 * D_, acc3);
}

// ctx = s @ Wv + bv  (per-head WMMA GEMM, M=32 batches)
__global__ void k_ctx(const unsigned short* __restrict__ sbf,
                      const unsigned short* __restrict__ WvT,
                      const float* __restrict__ bv,
                      unsigned short* __restrict__ ctx) {
  int bx = blockIdx.x;                 // 64: head(4) x mt(2) x nt(8)
  int head = bx >> 4, rem = bx & 15;
  int mt = rem >> 3, nt = rem & 7;
  int lane = threadIdx.x;
  const unsigned short* A = sbf + head * D_ + (size_t)(mt * 16) * (H_ * D_);
  int ncolbase = head * DH_ + nt * 16;
  v8f c = wmma_loop(A, H_ * D_, WvT, D_, ncolbase, 16, lane);
  int col = ncolbase + (lane & 15);
  int half = lane >> 4;
  float bvv = bv[col];
  for (int i = 0; i < 8; ++i) {
    int row = mt * 16 + half * 8 + i;
    ctx[(size_t)row * D_ + col] = f2bf(c[i] + bvv);
  }
}

// h1 = token + ctx @ Wo + bo
__global__ void k_h1(const unsigned short* __restrict__ ctx,
                     const unsigned short* __restrict__ WoT,
                     const float* __restrict__ token,
                     const float* __restrict__ bo,
                     float* __restrict__ h1) {
  int bx = blockIdx.x;                 // 64: mt(2) x nt(32)
  int mt = bx >> 5, nt = bx & 31;
  int lane = threadIdx.x;
  const unsigned short* A = ctx + (size_t)(mt * 16) * D_;
  v8f c = wmma_loop(A, D_, WoT, D_, nt * 16, 16, lane);
  int col = nt * 16 + (lane & 15);
  int half = lane >> 4;
  float add = token[col] + bo[col];
  for (int i = 0; i < 8; ++i) {
    int row = mt * 16 + half * 8 + i;
    h1[(size_t)row * D_ + col] = c[i] + add;
  }
}

__global__ void k_ln2(const float* __restrict__ h1,
                      const float* __restrict__ s, const float* __restrict__ bta,
                      unsigned short* __restrict__ h2) {
  __shared__ float red[256];
  int b = blockIdx.x, t = threadIdx.x;
  const float* r = h1 + (size_t)b * D_;
  float a0 = r[t], a1 = r[t + 256];
  float sx = block_reduce_sum(a0 + a1, red);
  float ss = block_reduce_sum(a0 * a0 + a1 * a1, red);
  float mean = sx * (1.0f / D_);
  float var  = ss * (1.0f / D_) - mean * mean;
  float rstd = rsqrtf(var + LN_EPS);
  h2[(size_t)b * D_ + t]       = f2bf((a0 - mean) * rstd * s[t]       + bta[t]);
  h2[(size_t)b * D_ + t + 256] = f2bf((a1 - mean) * rstd * s[t + 256] + bta[t + 256]);
}

// g = gelu_tanh(h2 @ W1 + b1)
__global__ void k_ffn1(const unsigned short* __restrict__ h2,
                       const unsigned short* __restrict__ W1T,
                       const float* __restrict__ b1,
                       unsigned short* __restrict__ g) {
  int bx = blockIdx.x;                 // 256: mt(2) x nt(128)
  int mt = bx >> 7, nt = bx & 127;
  int lane = threadIdx.x;
  const unsigned short* A = h2 + (size_t)(mt * 16) * D_;
  v8f c = wmma_loop(A, D_, W1T, D_, nt * 16, 16, lane);
  int col = nt * 16 + (lane & 15);
  int half = lane >> 4;
  float bb = b1[col];
  for (int i = 0; i < 8; ++i) {
    int row = mt * 16 + half * 8 + i;
    float xv = c[i] + bb;
    float gv = 0.5f * xv * (1.0f + tanhf(0.7978845608028654f * (xv + 0.044715f * xv * xv * xv)));
    g[(size_t)row * FF_ + col] = f2bf(gv);
  }
}

// out = h1 + g @ W2 + b2
__global__ void k_ffn2(const unsigned short* __restrict__ g,
                       const unsigned short* __restrict__ W2T,
                       const float* __restrict__ b2,
                       const float* __restrict__ h1,
                       float* __restrict__ out) {
  int bx = blockIdx.x;                 // 64: mt(2) x nt(32)
  int mt = bx >> 5, nt = bx & 31;
  int lane = threadIdx.x;
  const unsigned short* A = g + (size_t)(mt * 16) * FF_;
  v8f c = wmma_loop(A, FF_, W2T, FF_, nt * 16, 64, lane);
  int col = nt * 16 + (lane & 15);
  int half = lane >> 4;
  float bb = b2[col];
  for (int i = 0; i < 8; ++i) {
    int row = mt * 16 + half * 8 + i;
    out[(size_t)row * D_ + col] = c[i] + bb + h1[(size_t)row * D_ + col];
  }
}

extern "C" void kernel_launch(void* const* d_in, const int* in_sizes, int n_in,
                              void* d_out, int out_size, void* d_ws, size_t ws_size,
                              hipStream_t stream) {
  (void)in_sizes; (void)n_in; (void)out_size; (void)ws_size;
  const float* x     = (const float*)d_in[0];
  const int*   mask  = (const int*)d_in[1];
  const float* token = (const float*)d_in[2];
  const float* Wq    = (const float*)d_in[3];
  const float* bq    = (const float*)d_in[4];
  const float* Wk    = (const float*)d_in[5];
  const float* bk    = (const float*)d_in[6];
  const float* Wv    = (const float*)d_in[7];
  const float* bv    = (const float*)d_in[8];
  const float* Wo    = (const float*)d_in[9];
  const float* bo    = (const float*)d_in[10];
  const float* ln1s  = (const float*)d_in[11];
  const float* ln1b  = (const float*)d_in[12];
  const float* ln2s  = (const float*)d_in[13];
  const float* ln2b  = (const float*)d_in[14];
  const float* W1    = (const float*)d_in[15];
  const float* b1    = (const float*)d_in[16];
  const float* W2    = (const float*)d_in[17];
  const float* b2    = (const float*)d_in[18];
  float* out = (float*)d_out;

  char* ws = (char*)d_ws;
  size_t off = 0;
  auto alloc = [&](size_t bytes) -> void* {
    void* p = ws + off;
    off = (off + bytes + 255) & ~(size_t)255;
    return p;
  };

  unsigned short* h     = (unsigned short*)alloc((size_t)B_ * NPAD * D_ * 2);
  unsigned short* mbf   = (unsigned short*)alloc((size_t)16 * D_ * 2);  // padded to 16 cols
  float*          qb    = (float*)         alloc((size_t)H_ * 4);
  unsigned short* hclsb = (unsigned short*)alloc((size_t)D_ * 2);
  float*          sc    = (float*)         alloc((size_t)B_ * H_ * NPAD * 4);
  float*          s32   = (float*)         alloc((size_t)B_ * H_ * D_ * 4);
  unsigned short* sbf   = (unsigned short*)alloc((size_t)B_ * H_ * D_ * 2);
  unsigned short* WvT   = (unsigned short*)alloc((size_t)D_ * D_ * 2);
  unsigned short* WoT   = (unsigned short*)alloc((size_t)D_ * D_ * 2);
  unsigned short* W1T   = (unsigned short*)alloc((size_t)FF_ * D_ * 2);
  unsigned short* W2T   = (unsigned short*)alloc((size_t)D_ * FF_ * 2);
  unsigned short* ctx   = (unsigned short*)alloc((size_t)B_ * D_ * 2);
  float*          h1    = (float*)         alloc((size_t)B_ * D_ * 4);
  unsigned short* h2    = (unsigned short*)alloc((size_t)B_ * D_ * 2);
  unsigned short* gbuf  = (unsigned short*)alloc((size_t)B_ * FF_ * 2);

  dim3 tblk(32, 8);
  // weight converts (column-major bf16 for WMMA B-fragments)
  k_convT<<<dim3(D_ / 32, D_ / 32), tblk, 0, stream>>>(Wv, WvT, D_, D_);
  k_convT<<<dim3(D_ / 32, D_ / 32), tblk, 0, stream>>>(Wo, WoT, D_, D_);
  k_convT<<<dim3(FF_ / 32, D_ / 32), tblk, 0, stream>>>(W1, W1T, D_, FF_);
  k_convT<<<dim3(D_ / 32, FF_ / 32), tblk, 0, stream>>>(W2, W2T, FF_, D_);

  k_norm_ln1<<<B_ * N_, 256, 0, stream>>>(x, ln1s, ln1b, h);
  hipMemsetAsync(mbf, 0, (size_t)16 * D_ * 2, stream);   // zero pad cols 4..15
  k_prep_small<<<1, 256, 0, stream>>>(token, ln1s, ln1b, Wq, bq, Wk, bk, hclsb, mbf, qb);
  k_fill_pad<<<B_ * 32, 128, 0, stream>>>(h, hclsb);

  k_scores<<<B_ * 34, 32, 0, stream>>>(h, mbf, qb, sc);
  k_softmax<<<B_ * H_, 256, 0, stream>>>(sc, mask);

  hipMemsetAsync(s32, 0, (size_t)B_ * H_ * D_ * 4, stream);
  k_wsum<<<B_ * 8, 256, 0, stream>>>(h, sc, s32);
  k_conv<<<64, 256, 0, stream>>>(s32, sbf, B_ * H_ * D_);

  k_ctx<<<64, 32, 0, stream>>>(sbf, WvT, bv, ctx);
  k_h1<<<64, 32, 0, stream>>>(ctx, WoT, token, bo, h1);
  k_ln2<<<B_, 256, 0, stream>>>(h1, ln2s, ln2b, h2);
  k_ffn1<<<256, 32, 0, stream>>>(h2, W1T, b1, gbuf);
  k_ffn2<<<64, 32, 0, stream>>>(gbuf, W2T, b2, h1, out);
}
